// CausalSelfAttention_12558484374117
// MI455X (gfx1250) — matrix operand order repaired and checked by reading, not compiled, not run on hardware
//
#include <hip/hip_runtime.h>

typedef __attribute__((ext_vector_type(8)))  _Float16 v8h;
typedef __attribute__((ext_vector_type(16))) _Float16 v16h;
typedef __attribute__((ext_vector_type(8)))  float    v8f;

#define SHUF16(lo, hi) __builtin_shufflevector((lo), (hi), 0,1,2,3,4,5,6,7,8,9,10,11,12,13,14,15)

// Async global->LDS copy (CDNA5, tracked by ASYNCcnt). VDST = LDS byte offset,
// VADDR = 64-bit global address, no SADDR.
__device__ __forceinline__ void async_b128(unsigned lds_off, const _Float16* g) {
    asm volatile("global_load_async_to_lds_b128 %0, %1, off"
                 :: "v"(lds_off), "v"(g) : "memory");
}

// ---------------------------------------------------------------------------
// f32 -> f16 convert
// ---------------------------------------------------------------------------
__global__ void cvt_f32_to_f16(const float* __restrict__ in, _Float16* __restrict__ out, size_t n) {
    size_t i = (size_t)blockIdx.x * blockDim.x + threadIdx.x;
    if (i < n) out[i] = (_Float16)in[i];
}

// ---------------------------------------------------------------------------
// Tiled WMMA GEMM:  out[M][N] = A[M][K] * Bm[N][K]^T + bias[N]
// BM=128, BN=64, BK=32; 256 threads = 8 waves arranged 4(M) x 2(N),
// each wave computes 2x2 tiles of 16x16. Double-buffered LDS filled by
// async-to-LDS copies, pipelined with s_wait_asynccnt.
// ---------------------------------------------------------------------------
__global__ __launch_bounds__(256) void gemm_wmma_f16(
    const _Float16* __restrict__ A,   // [M][K] row-major, f16
    const _Float16* __restrict__ Bm,  // [N][K] row-major (i.e. B^T), f16
    const float*    __restrict__ bias,// [N]
    float*          __restrict__ Cf,  // optional f32 out [M][N]
    _Float16*       __restrict__ Ch,  // optional f16 out [M][N]
    int M, int N, int K)
{
    constexpr int LDA = 40;                  // padded halves per LDS row (80B, conflict-free)
    __shared__ _Float16 lA[2][128 * LDA];
    __shared__ _Float16 lB[2][64 * LDA];

    const int tid = threadIdx.x;
    const int wid = tid >> 5, lid = tid & 31;
    const int laneRow = lid & 15, sel = lid >> 4;
    const int m0 = blockIdx.y * 128;
    const int n0 = blockIdx.x * 64;
    const int wm = wid >> 1;   // 0..3
    const int wn = wid & 1;    // 0..1

    // Per-thread async chunk coords (1 chunk = b128 = 8 halves).
    // A tile: 128 rows x 32 halves = 512 chunks -> 2 per thread.
    // B tile:  64 rows x 32 halves = 256 chunks -> 1 per thread.
    const int a0r = tid >> 2;            // rows 0..63
    const int a1r = 64 + (tid >> 2);     // rows 64..127
    const int ac  = (tid & 3) * 8;       // half-offset within row
    const int br  = tid >> 2;            // rows 0..63
    const int bc  = ac;

    const int S = K / 32;

    auto prefetch = [&](int stage, int buf) {
        const int k0 = stage * 32;
        async_b128((unsigned)(uintptr_t)&lA[buf][a0r * LDA + ac],
                   A + (size_t)(m0 + a0r) * K + k0 + ac);
        async_b128((unsigned)(uintptr_t)&lA[buf][a1r * LDA + ac],
                   A + (size_t)(m0 + a1r) * K + k0 + ac);
        async_b128((unsigned)(uintptr_t)&lB[buf][br * LDA + bc],
                   Bm + (size_t)(n0 + br) * K + k0 + bc);
    };

    v8f acc[2][2] = {};
    prefetch(0, 0);

    for (int s = 0; s < S; ++s) {
        const int buf = s & 1;
        if (s + 1 < S) {
            prefetch(s + 1, buf ^ 1);
            // current stage = first 3 outstanding (per-wave in-order completion)
            asm volatile("s_wait_asynccnt 0x3" ::: "memory");
        } else {
            asm volatile("s_wait_asynccnt 0x0" ::: "memory");
        }
        __syncthreads();   // all waves' portions landed

        v16h af[2], bf[2];
        #pragma unroll
        for (int t = 0; t < 2; ++t) {
            // A-fragment: interleaved K pattern {sel*8+0..7, 16+sel*8+0..7}
            const int r = wm * 32 + t * 16 + laneRow;
            v8h lo = *(const v8h*)(&lA[buf][r * LDA + sel * 8]);
            v8h hi = *(const v8h*)(&lA[buf][r * LDA + 16 + sel * 8]);
            af[t] = SHUF16(lo, hi);
        }
        #pragma unroll
        for (int t = 0; t < 2; ++t) {
            // B-fragment: interleaved K pattern {sel*8+0..7, 16+sel*8+0..7}
            const int r = wn * 32 + t * 16 + laneRow;
            v8h lo = *(const v8h*)(&lB[buf][r * LDA + sel * 8]);
            v8h hi = *(const v8h*)(&lB[buf][r * LDA + 16 + sel * 8]);
            bf[t] = SHUF16(lo, hi);
        }
        #pragma unroll
        for (int i = 0; i < 2; ++i)
            #pragma unroll
            for (int t = 0; t < 2; ++t)
                acc[i][t] = __builtin_amdgcn_wmma_f32_16x16x32_f16(
                    false, af[i], false, bf[t], (short)0, acc[i][t], false, false);

        __syncthreads();   // done reading buf before it is refilled at s+2
    }

    // epilogue: C/D layout -> memory (+bias); output-format branch hoisted
    #pragma unroll
    for (int i = 0; i < 2; ++i) {
        #pragma unroll
        for (int t = 0; t < 2; ++t) {
            const int n = n0 + wn * 32 + t * 16 + laneRow;
            const float bv = bias[n];
            if (Ch) {
                #pragma unroll
                for (int j = 0; j < 8; ++j) {
                    const int m = m0 + wm * 32 + i * 16 + j + 8 * sel;
                    Ch[(size_t)m * N + n] = (_Float16)(acc[i][t][j] + bv);
                }
            } else {
                #pragma unroll
                for (int j = 0; j < 8; ++j) {
                    const int m = m0 + wm * 32 + i * 16 + j + 8 * sel;
                    Cf[(size_t)m * N + n] = acc[i][t][j] + bv;
                }
            }
        }
    }
}

// ---------------------------------------------------------------------------
// RoPE + split qkv[M][3C] into q,k [BH][T][D] (rotated) and vT [BH][D][T]
// one thread per (b,h,t,d-pair); D=64 -> 32 pairs
// ---------------------------------------------------------------------------
__global__ void rope_split(const _Float16* __restrict__ qkv,
                           _Float16* __restrict__ qh,
                           _Float16* __restrict__ kh,
                           _Float16* __restrict__ vT)
{
    constexpr int T = 2048, D = 64, C3 = 3072;
    int idx = blockIdx.x * blockDim.x + threadIdx.x;
    if (idx >= 4 * 16 * T * 32) return;
    const int d2 = idx & 31;
    const int t  = (idx >> 5) & (T - 1);
    const int bh = idx >> 16;          // 0..63
    const int b  = bh >> 4, h = bh & 15;

    const size_t base = ((size_t)(b * T + t)) * C3 + h * 192;
    const float q0 = (float)qkv[base + 2 * d2];
    const float q1 = (float)qkv[base + 2 * d2 + 1];
    const float k0 = (float)qkv[base + 64 + 2 * d2];
    const float k1 = (float)qkv[base + 64 + 2 * d2 + 1];
    const float v0 = (float)qkv[base + 128 + 2 * d2];
    const float v1 = (float)qkv[base + 128 + 2 * d2 + 1];

    // theta_i = 10000^(-d2/32) = exp2(-d2 * log2(10000)/32)
    const float theta = exp2f(-(float)d2 * (13.2877123795494f / 32.0f));
    const float ang = (float)t * theta;
    const float sn = __sinf(ang), cs = __cosf(ang);

    const size_t qi = (((size_t)bh * T + t)) * D + 2 * d2;
    qh[qi]     = (_Float16)(q0 * cs - q1 * sn);
    qh[qi + 1] = (_Float16)(q0 * sn + q1 * cs);
    kh[qi]     = (_Float16)(k0 * cs - k1 * sn);
    kh[qi + 1] = (_Float16)(k0 * sn + k1 * cs);

    const size_t vi = ((size_t)bh * D + 2 * d2) * T + t;
    vT[vi]     = (_Float16)v0;
    vT[vi + T] = (_Float16)v1;
}

// ---------------------------------------------------------------------------
// Causal flash attention. Block = 128 threads = 4 waves, each wave owns a
// 16-row q strip. grid = (T/64, B*H).
// ---------------------------------------------------------------------------
__global__ __launch_bounds__(128) void attn_wmma(
    const _Float16* __restrict__ Qh,  // [BH][T][D]
    const _Float16* __restrict__ Kh,  // [BH][T][D]
    const _Float16* __restrict__ VT,  // [BH][D][T]
    _Float16*       __restrict__ Oh)  // [B][T][H*D] f16 (input to out-proj)
{
    constexpr int T = 2048, D = 64;
    __shared__ _Float16 pst[4][16 * 40];   // per-wave P staging (C-layout -> A-layout)

    const int tid = threadIdx.x;
    const int w = tid >> 5, lid = tid & 31;
    const int laneRow = lid & 15, sel = lid >> 4;
    const int bh = blockIdx.y;
    const int q0 = blockIdx.x * 64 + w * 16;

    const _Float16* Qp = Qh + (size_t)bh * T * D;
    const _Float16* Kp = Kh + (size_t)bh * T * D;
    const _Float16* Vp = VT + (size_t)bh * D * T;

    // Q A-fragments (two d-steps of 32)
    v16h qf[2];
    {
        const int r = q0 + laneRow;
        #pragma unroll
        for (int ds = 0; ds < 2; ++ds) {
            v8h lo = *(const v8h*)(Qp + (size_t)r * D + ds * 32 + sel * 8);
            v8h hi = *(const v8h*)(Qp + (size_t)r * D + ds * 32 + 16 + sel * 8);
            qf[ds] = SHUF16(lo, hi);
        }
    }

    v8f o[4] = {};
    float mrun[8], lrun[8];
    #pragma unroll
    for (int j = 0; j < 8; ++j) { mrun[j] = -1e30f; lrun[j] = 0.0f; }

    for (int kt0 = 0; kt0 <= q0 + 15; kt0 += 32) {
        // scores: two 16x16 tiles over 32 kt columns, contracting D=64
        v8f s[2] = {};
        #pragma unroll
        for (int c = 0; c < 2; ++c) {
            const int kr = kt0 + c * 16 + laneRow;   // B column = K row (contiguous)
            #pragma unroll
            for (int ds = 0; ds < 2; ++ds) {
                v8h lo = *(const v8h*)(Kp + (size_t)kr * D + ds * 32 + sel * 8);
                v8h hi = *(const v8h*)(Kp + (size_t)kr * D + ds * 32 + 16 + sel * 8);
                v16h kf = SHUF16(lo, hi);
                s[c] = __builtin_amdgcn_wmma_f32_16x16x32_f16(
                    false, qf[ds], false, kf, (short)0, s[c], false, false);
            }
        }

        // scale + causal mask + per-row max (butterfly within 16-lane halves)
        const bool needMask = (kt0 + 31 > q0);
        float rm[8];
        #pragma unroll
        for (int j = 0; j < 8; ++j) {
            const int row = q0 + j + 8 * sel;
            float s0 = s[0][j] * 0.125f;
            float s1 = s[1][j] * 0.125f;
            if (needMask) {
                if (kt0 + laneRow > row)      s0 = -1e30f;
                if (kt0 + 16 + laneRow > row) s1 = -1e30f;
            }
            s[0][j] = s0; s[1][j] = s1;
            float v = fmaxf(s0, s1);
            v = fmaxf(v, __shfl_xor(v, 1, 32));
            v = fmaxf(v, __shfl_xor(v, 2, 32));
            v = fmaxf(v, __shfl_xor(v, 4, 32));
            v = fmaxf(v, __shfl_xor(v, 8, 32));
            rm[j] = v;
        }

        // online softmax update
        float alpha[8];
        #pragma unroll
        for (int j = 0; j < 8; ++j) {
            const float mnew = fmaxf(mrun[j], rm[j]);
            alpha[j] = __expf(mrun[j] - mnew);
            mrun[j] = mnew;
            const float p0 = __expf(s[0][j] - mnew);
            const float p1 = __expf(s[1][j] - mnew);
            s[0][j] = p0; s[1][j] = p1;
            float v = p0 + p1;
            v += __shfl_xor(v, 1, 32);
            v += __shfl_xor(v, 2, 32);
            v += __shfl_xor(v, 4, 32);
            v += __shfl_xor(v, 8, 32);
            lrun[j] = lrun[j] * alpha[j] + v;
        }
        #pragma unroll
        for (int c = 0; c < 4; ++c)
            #pragma unroll
            for (int j = 0; j < 8; ++j) o[c][j] *= alpha[j];

        // P: C-layout -> A-layout through per-wave LDS tile
        #pragma unroll
        for (int j = 0; j < 8; ++j) {
            const int r = j + 8 * sel;
            pst[w][r * 40 + laneRow]      = (_Float16)s[0][j];
            pst[w][r * 40 + 16 + laneRow] = (_Float16)s[1][j];
        }
        __asm__ volatile("" ::: "memory");  // keep DS loads after DS stores (same-wave, HW in-order)
        v16h pf;
        {
            v8h lo = *(const v8h*)(&pst[w][laneRow * 40 + sel * 8]);
            v8h hi = *(const v8h*)(&pst[w][laneRow * 40 + 16 + sel * 8]);
            pf = SHUF16(lo, hi);
        }

        // O += P (16x32) * V (32x64), 4 d-tiles; vT rows contiguous in t
        #pragma unroll
        for (int c2 = 0; c2 < 4; ++c2) {
            const int d = c2 * 16 + laneRow;
            v8h lo = *(const v8h*)(Vp + (size_t)d * T + kt0 + sel * 8);
            v8h hi = *(const v8h*)(Vp + (size_t)d * T + kt0 + 16 + sel * 8);
            v16h vf = SHUF16(lo, hi);
            o[c2] = __builtin_amdgcn_wmma_f32_16x16x32_f16(
                false, pf, false, vf, (short)0, o[c2], false, false);
        }
    }

    // normalize and write [b][t][h*64+d]
    const int b = bh >> 4, h = bh & 15;
    #pragma unroll
    for (int j = 0; j < 8; ++j) {
        const float inv = 1.0f / lrun[j];
        const int trow = q0 + j + 8 * sel;
        const size_t base = ((size_t)(b * T + trow)) * 1024 + h * 64;
        #pragma unroll
        for (int c2 = 0; c2 < 4; ++c2)
            Oh[base + c2 * 16 + laneRow] = (_Float16)(o[c2][j] * inv);
    }
}

// ---------------------------------------------------------------------------
extern "C" void kernel_launch(void* const* d_in, const int* in_sizes, int n_in,
                              void* d_out, int out_size, void* d_ws, size_t ws_size,
                              hipStream_t stream) {
    const float* x     = (const float*)d_in[0];
    const float* w_qkv = (const float*)d_in[1];
    const float* b_qkv = (const float*)d_in[2];
    const float* w_out = (const float*)d_in[3];
    const float* b_out = (const float*)d_in[4];
    float* out = (float*)d_out;

    constexpr int Bq = 4, T = 2048, C = 1024, H = 16;
    constexpr int M = Bq * T;          // 8192
    constexpr int C3 = 3 * C;          // 3072

    char* ws = (char*)d_ws;
    auto alloc = [&](size_t bytes) { char* p = ws; ws += (bytes + 255) & ~(size_t)255; return p; };
    _Float16* xh    = (_Float16*)alloc((size_t)M * C * 2);
    _Float16* wqkvh = (_Float16*)alloc((size_t)C3 * C * 2);
    _Float16* wouth = (_Float16*)alloc((size_t)C * C * 2);
    _Float16* qkvh  = (_Float16*)alloc((size_t)M * C3 * 2);
    _Float16* qh    = (_Float16*)alloc((size_t)M * C * 2);   // [BH][T][64]
    _Float16* kh    = (_Float16*)alloc((size_t)M * C * 2);
    _Float16* vt    = (_Float16*)alloc((size_t)M * C * 2);   // [BH][64][T]
    _Float16* ah    = (_Float16*)alloc((size_t)M * C * 2);

    // converts
    cvt_f32_to_f16<<<(M * C + 255) / 256, 256, 0, stream>>>(x, xh, (size_t)M * C);
    cvt_f32_to_f16<<<(C3 * C + 255) / 256, 256, 0, stream>>>(w_qkv, wqkvh, (size_t)C3 * C);
    cvt_f32_to_f16<<<(C * C + 255) / 256, 256, 0, stream>>>(w_out, wouth, (size_t)C * C);

    // qkv = x @ w_qkv^T + b_qkv   (f16 out)
    gemm_wmma_f16<<<dim3(C3 / 64, M / 128), 256, 0, stream>>>(
        xh, wqkvh, b_qkv, nullptr, qkvh, M, C3, C);

    // rope + split + V transpose
    rope_split<<<(Bq * H * T * 32) / 256, 256, 0, stream>>>(qkvh, qh, kh, vt);

    // causal attention
    attn_wmma<<<dim3(T / 64, Bq * H), 128, 0, stream>>>(qh, kh, vt, ah);

    // y = attn @ w_out^T + b_out  (f32 out)
    gemm_wmma_f16<<<dim3(C / 64, M / 128), 256, 0, stream>>>(
        ah, wouth, b_out, out, nullptr, M, C, C);
}
